// BinaryLinear_49838800503279
// MI455X (gfx1250) — matrix operand order repaired and checked by reading, not compiled, not run on hardware
//
#include <hip/hip_runtime.h>
#include <stdint.h>

// ---------------------------------------------------------------------------
// BinaryLinear: out[m,n] = sum_k x[m,k] * (aa*clamp(kk*w[n,k],-1,1)) + bias[n]
// M=32768, N=1024, K=1024.  bf16 WMMA (16x16x32) with f32 accumulation.
// Weight tile staged via TDM tensor_load_to_lds (TENSORcnt) when available,
// else GLOBAL_LOAD_ASYNC_TO_LDS (ASYNCcnt), else plain loads.
// ---------------------------------------------------------------------------

typedef __attribute__((ext_vector_type(16))) __bf16 v16bf;
typedef __attribute__((ext_vector_type(8)))  __bf16 v8bf;
typedef __attribute__((ext_vector_type(8)))  float  v8f;

// Async-to-LDS payload pointers (prototype confirmed by round-2 diagnostic):
typedef int v4i_ __attribute__((vector_size(4 * sizeof(int))));
typedef __attribute__((address_space(1))) v4i_* async_gptr;
typedef __attribute__((address_space(3))) v4i_* async_lptr;

// TDM descriptor group types (per probe doc):
typedef unsigned int u32x4 __attribute__((ext_vector_type(4)));
typedef int          i32x4 __attribute__((ext_vector_type(4)));
typedef int          i32x8 __attribute__((ext_vector_type(8)));

#if defined(__HIP_DEVICE_COMPILE__)
# if __has_builtin(__builtin_amdgcn_global_load_async_to_lds_b128)
#  define HAVE_ASYNC_LDS 1
# endif
# if __has_builtin(__builtin_amdgcn_cvt_pk_bf16_f32)
#  define HAVE_CVT_PK 1
# endif
# if __has_builtin(__builtin_amdgcn_tensor_load_to_lds) && \
     __has_builtin(__builtin_amdgcn_s_wait_tensorcnt)
#  define HAVE_TDM 1
# endif
#endif

#define M_TOTAL 32768
#define N_TOTAL 1024
#define K_TOTAL 1024
#define BM 128
#define BN 128
#define BK 32
#define KTILES (K_TOTAL / BK)   // 32
#define LDA (BK + 8)            // 40 halves = 80 B row stride (16B-aligned, pad vs bank conflicts)
#define LDB (BK + 8)

__device__ __forceinline__ uint16_t f32_to_bf16_rne(float f) {
  uint32_t u = __float_as_uint(f);
  uint32_t r = u + 0x7FFFu + ((u >> 16) & 1u);   // round-to-nearest-even
  return (uint16_t)(r >> 16);
}

__device__ __forceinline__ uint32_t pack2_bf16(float a, float b) {
#ifdef HAVE_CVT_PK
  auto v = __builtin_amdgcn_cvt_pk_bf16_f32(a, b);
  uint32_t u;
  __builtin_memcpy(&u, &v, sizeof(u));
  return u;
#else
  return (uint32_t)f32_to_bf16_rne(a) | ((uint32_t)f32_to_bf16_rne(b) << 16);
#endif
}

// ---------------- Kernel 1: weight binarization -> bf16 ---------------------
__global__ __launch_bounds__(256)
void binlin_prep(const float* __restrict__ w,
                 const float* __restrict__ kkp,
                 const float* __restrict__ aap,
                 uint16_t* __restrict__ wbin) {
  const int idx = blockIdx.x * 256 + threadIdx.x;     // one float4 per thread
  const float kk = kkp[0];
  const float aa = aap[0];
  const float4 v = ((const float4*)w)[idx];
  const float b0 = aa * fminf(1.0f, fmaxf(-1.0f, kk * v.x));
  const float b1 = aa * fminf(1.0f, fmaxf(-1.0f, kk * v.y));
  const float b2 = aa * fminf(1.0f, fmaxf(-1.0f, kk * v.z));
  const float b3 = aa * fminf(1.0f, fmaxf(-1.0f, kk * v.w));
  ((uint2*)wbin)[idx] = make_uint2(pack2_bf16(b0, b1), pack2_bf16(b2, b3));
}

// ---------------- staging helpers ------------------------------------------
// A tile: 128x32 fp32 -> bf16 via registers (conversion forces the VGPR path).
__device__ __forceinline__ void stage_A(const float* __restrict__ x,
                                        uint16_t (*__restrict__ Asb)[LDA],
                                        int t, int m0, int k0) {
#pragma unroll
  for (int i = 0; i < 4; ++i) {
    const int c   = t + 256 * i;
    const int row = c >> 3;            // 8 float4 per 32-wide row
    const int col = (c & 7) * 4;
    const float4 v = *(const float4*)(x + (size_t)(m0 + row) * K_TOTAL + k0 + col);
    *(uint2*)&Asb[row][col] = make_uint2(pack2_bf16(v.x, v.y), pack2_bf16(v.z, v.w));
  }
}

#ifdef HAVE_TDM
// B tile via Tensor Data Mover: one instruction moves the whole 128x32 bf16
// tile, with HW LDS padding reproducing the 80B padded row stride.
// D# built per ISA 08_async_tensor §8.3/8.4.
__device__ __forceinline__ void tdm_stage_B(const uint16_t* g_tile, uint32_t lds_off) {
  const uint64_t ga = (uint64_t)(uintptr_t)g_tile;
  u32x4 g0;
  g0[0] = 1u;                                  // count=1, user descriptor
  g0[1] = lds_off;                             // lds_addr [63:32]
  g0[2] = (uint32_t)ga;                        // global_addr [95:64]
  g0[3] = (uint32_t)((ga >> 32) & 0x01FFFFFFu) // global_addr [120:96]
        | (2u << 30);                          // type=2 ("image")

  i32x8 g1;
  g1[0] = (1 << 16)                            // data_size = 2 bytes
        | (1 << 20)                            // pad_enable (load only)
        | (3 << 22)                            // pad_interval: 16 DWORDs (= one 64B row)
        | (3 << 25);                           // pad_amount:   4 DWORDs (= 16B pad)
  g1[1] = (int)(K_TOTAL << 16);                // tensor_dim0[15:0] @ [63:48]
  g1[2] = (int)((unsigned)(K_TOTAL >> 16)      // tensor_dim0[31:16]
        | ((N_TOTAL & 0xFFFFu) << 16));        // tensor_dim1[15:0]
  g1[3] = (int)((unsigned)(N_TOTAL >> 16)      // tensor_dim1[31:16]
        | ((unsigned)BK << 16));               // tile_dim0 = 32
  g1[4] = (int)BN;                             // tile_dim1 = 128 (tile_dim2 = 0)
  g1[5] = (int)K_TOTAL;                        // tensor_dim0_stride[31:0]
  g1[6] = 0;                                   // stride hi / dim1_stride lo
  g1[7] = 0;

  const i32x4 gz = {0, 0, 0, 0};               // groups 2/3 unused (2-D tensor)
#if __clang_major__ >= 23
  const i32x8 gz8 = {0, 0, 0, 0, 0, 0, 0, 0};
  __builtin_amdgcn_tensor_load_to_lds(g0, g1, gz, gz, gz8, 0);
#else
  __builtin_amdgcn_tensor_load_to_lds(g0, g1, gz, gz, 0);
#endif
}
#endif

// Fallback B staging: per-lane async DMA to LDS (or plain loads).
__device__ __forceinline__ void stage_B(const uint16_t* __restrict__ wbin,
                                        uint16_t (*__restrict__ Bsb)[LDB],
                                        int t, int n0, int k0) {
#pragma unroll
  for (int i = 0; i < 2; ++i) {
    const int c   = t + 256 * i;
    const int row = c >> 2;
    const int col = (c & 3) * 8;
    const uint16_t* g = wbin + (size_t)(n0 + row) * K_TOTAL + k0 + col;
    uint16_t* l = &Bsb[row][col];
#ifdef HAVE_ASYNC_LDS
    __builtin_amdgcn_global_load_async_to_lds_b128(
        (async_gptr)(uintptr_t)g, (async_lptr)(uint32_t)(uintptr_t)l, 0, 0);
#else
    *(uint4*)l = *(const uint4*)g;
#endif
  }
}

__device__ __forceinline__ void wait_async_lds() {
#ifdef HAVE_ASYNC_LDS
  asm volatile("s_wait_asynccnt 0" ::: "memory");
#endif
}

// Issue B staging for one K-step; wave 0 owns the TDM descriptor.
__device__ __forceinline__ void issue_B(const uint16_t* __restrict__ wbin,
                                        uint16_t (*__restrict__ Bsb)[LDB],
                                        int t, int wave, int n0, int k0) {
#ifdef HAVE_TDM
  if (wave == 0)
    tdm_stage_B(wbin + (size_t)n0 * K_TOTAL + k0,
                (uint32_t)(uintptr_t)&Bsb[0][0]);
#else
  stage_B(wbin, Bsb, t, n0, k0);
#endif
}

__device__ __forceinline__ void wait_B(int wave) {
#ifdef HAVE_TDM
  if (wave == 0) __builtin_amdgcn_s_wait_tensorcnt(0);
#else
  wait_async_lds();
#endif
}

// ---------------- Kernel 2: tiled bf16 WMMA GEMM ----------------------------
__global__ __launch_bounds__(256)
void binlin_gemm(const float* __restrict__ x,
                 const uint16_t* __restrict__ wbin,
                 const float* __restrict__ bias,
                 float* __restrict__ out) {
  __shared__ __align__(16) uint16_t As[2][BM][LDA];  // 20 KB
  __shared__ __align__(16) uint16_t Bs[2][BN][LDB];  // 20 KB
  __shared__ __align__(16) float    BiasS[BN];       // 512 B

  const int t    = threadIdx.x;
  const int m0   = blockIdx.y * BM;
  const int n0   = blockIdx.x * BN;
  const int lane = t & 31;
  const int wave = t >> 5;
  const int wm   = (wave & 3) * 32;
  const int wn   = (wave >> 2) * 64;
  const int l16  = lane & 15;
  const int hi   = lane >> 4;

  v8f acc[2][4];
#pragma unroll
  for (int i = 0; i < 2; ++i)
#pragma unroll
    for (int j = 0; j < 4; ++j)
      acc[i][j] = (v8f){};

  // Bias slice for this block -> LDS (async DMA, wave 0: 32 lanes x 16B).
#ifdef HAVE_ASYNC_LDS
  if (t < 32)
    __builtin_amdgcn_global_load_async_to_lds_b128(
        (async_gptr)(uintptr_t)(bias + n0 + t * 4),
        (async_lptr)(uint32_t)(uintptr_t)&BiasS[t * 4], 0, 0);
#else
  if (t < BN) BiasS[t] = bias[n0 + t];
#endif

  issue_B(wbin, Bs[0], t, wave, n0, 0);
  stage_A(x, As[0], t, m0, 0);
  wait_B(wave);
  wait_async_lds();
  __syncthreads();

  for (int kt = 0; kt < KTILES; ++kt) {
    const int buf = kt & 1;
    if (kt + 1 < KTILES) {
      issue_B(wbin, Bs[buf ^ 1], t, wave, n0, (kt + 1) * BK);   // DMA first
      stage_A(x, As[buf ^ 1], t, m0, (kt + 1) * BK);
    }
    if (kt + 2 < KTILES)   // warm L2/WGP$ for the x stream two tiles ahead
      __builtin_prefetch(x + (size_t)(m0 + (t >> 1)) * K_TOTAL + (kt + 2) * BK, 0, 1);

    // A fragments: 16-bit A 16x32 layout — lanes 0-15 hold K {0..7,16..23},
    // lanes 16-31 hold K {8..15,24..31} for rows M=0..15.
    v16bf a[2];
#pragma unroll
    for (int i = 0; i < 2; ++i) {
      const uint16_t* p = &As[buf][wm + i * 16 + l16][hi * 8];
      v8bf lo = *(const v8bf*)p;
      v8bf hh = *(const v8bf*)(p + 16);
      a[i] = __builtin_shufflevector(lo, hh, 0, 1, 2, 3, 4, 5, 6, 7,
                                             8, 9, 10, 11, 12, 13, 14, 15);
    }
    // B fragments: 16-bit B 32x16 layout — lanes 0-15 hold K {0..7,16..23},
    // lanes 16-31 hold K {8..15,24..31} for column N = lane%16.  B[k][n] = wbin[n][k].
    v16bf b[4];
#pragma unroll
    for (int j = 0; j < 4; ++j) {
      const uint16_t* p = &Bs[buf][wn + j * 16 + l16][hi * 8];
      v8bf lo = *(const v8bf*)p;
      v8bf hh = *(const v8bf*)(p + 16);
      b[j] = __builtin_shufflevector(lo, hh, 0, 1, 2, 3, 4, 5, 6, 7,
                                             8, 9, 10, 11, 12, 13, 14, 15);
    }

#pragma unroll
    for (int i = 0; i < 2; ++i)
#pragma unroll
      for (int j = 0; j < 4; ++j)
        acc[i][j] = __builtin_amdgcn_wmma_f32_16x16x32_bf16(
            false, a[i], false, b[j], (short)0, acc[i][j], false, false);

    wait_B(wave);        // TENSORcnt/ASYNCcnt not covered by the barrier
    __syncthreads();
  }

  // Epilogue: C/D layout — VGPR r, lane L -> (M = (L/16)*8 + r, N = L%16).
#pragma unroll
  for (int j = 0; j < 4; ++j) {
    const int nt = n0 + wn + j * 16;
    const float bv = BiasS[wn + j * 16 + l16];
#pragma unroll
    for (int i = 0; i < 2; ++i) {
      const int mt = m0 + wm + i * 16 + hi * 8;
      float* op = out + (size_t)mt * N_TOTAL + nt + l16;
#pragma unroll
      for (int r = 0; r < 8; ++r)
        op[(size_t)r * N_TOTAL] = acc[i][j][r] + bv;
    }
  }
}

// ---------------------------------------------------------------------------
extern "C" void kernel_launch(void* const* d_in, const int* in_sizes, int n_in,
                              void* d_out, int out_size, void* d_ws, size_t ws_size,
                              hipStream_t stream) {
  const float* x    = (const float*)d_in[0];   // [4,8192,1024] fp32
  const float* w    = (const float*)d_in[1];   // [1024,1024]   fp32
  const float* bias = (const float*)d_in[2];   // [1024]        fp32
  const float* kk   = (const float*)d_in[3];   // scalar
  const float* aa   = (const float*)d_in[4];   // scalar
  float* out        = (float*)d_out;           // [4,8192,1024] fp32
  uint16_t* wbin    = (uint16_t*)d_ws;         // 2 MB bf16 binarized weight

  (void)in_sizes; (void)n_in; (void)out_size; (void)ws_size;

  binlin_prep<<<(N_TOTAL * K_TOTAL) / (256 * 4), 256, 0, stream>>>(w, kk, aa, wbin);

  dim3 grid(N_TOTAL / BN, M_TOTAL / BM);       // (8, 256) = 2048 workgroups
  binlin_gemm<<<grid, 256, 0, stream>>>(x, wbin, bias, out);
}
